// StaticCRF_65584150610167
// MI455X (gfx1250) — matrix operand order repaired and checked by reading, not compiled, not run on hardware
//
#include <hip/hip_runtime.h>

// Problem constants (from reference): x (B=16, C=16, H=384, W=384) f32,
// spatial_spacings (16, 2) f32. 5 mean-field iterations of
//   x_{t+1} = x0 + gauss_smooth(softmax_C(x_t))     (Potts compat = -I folds away)
#define B_   16
#define C_   16
#define H_   384
#define W_   384
#define HW_  (H_ * W_)
#define CHW_ (C_ * HW_)
#define NTOT_ ((long)B_ * CHW_)
#define WT_  (W_ / 16)
#define HT_  (H_ / 16)
#define TILES_ (B_ * C_ * HT_ * WT_)

typedef __attribute__((ext_vector_type(16))) _Float16 v16h;
typedef __attribute__((ext_vector_type(8)))  _Float16 v8h;
typedef __attribute__((ext_vector_type(4)))  _Float16 v4h;
typedef __attribute__((ext_vector_type(8)))  float    v8f;
typedef __attribute__((ext_vector_type(4)))  float    v4f;

// Gaussian tap for d in [0,10]; center (d==5) zeroed (message passing excludes
// self); zero outside band. kern[d] = exp(-((d-5)*spacing)^2/2), INV_THETA=1.
__device__ __forceinline__ float gauss_tap(int d, float s) {
  bool ok = (d >= 0) & (d <= 10) & (d != 5);
  float f = (float)(d - 5) * s;
  float v = __expf(-0.5f * f * f);
  return ok ? v : 0.0f;
}

// ---------------------------------------------------------------------------
// Kernel 1: channel softmax, f32 in -> f16 probabilities out. 4 consecutive
// pixels per thread: 16x global_load_b128 in, 16x global_store_b64 out.
// ---------------------------------------------------------------------------
__global__ __launch_bounds__(256) void crf_softmax_f16(
    const float* __restrict__ x, _Float16* __restrict__ p) {
  int q = blockIdx.x * 256 + threadIdx.x;             // quad-pixel index
  if (q >= (B_ * HW_) / 4) return;
  int b   = q / (HW_ / 4);
  int hw4 = q - b * (HW_ / 4);
  long base = (long)b * CHW_ + (long)hw4 * 4;

  v4f v[C_];
  float m[4] = {-3.4e38f, -3.4e38f, -3.4e38f, -3.4e38f};
#pragma unroll
  for (int c = 0; c < C_; ++c) {
    v[c] = *(const v4f*)(x + base + (long)c * HW_);
#pragma unroll
    for (int j = 0; j < 4; ++j) m[j] = fmaxf(m[j], v[c][j]);
  }
  float sum[4] = {0.f, 0.f, 0.f, 0.f};
#pragma unroll
  for (int c = 0; c < C_; ++c) {
#pragma unroll
    for (int j = 0; j < 4; ++j) {
      float e = __expf(v[c][j] - m[j]);
      v[c][j] = e;
      sum[j] += e;
    }
  }
  float inv[4];
#pragma unroll
  for (int j = 0; j < 4; ++j) inv[j] = 1.0f / sum[j];
#pragma unroll
  for (int c = 0; c < C_; ++c) {
    v4h h4;
#pragma unroll
    for (int j = 0; j < 4; ++j) h4[j] = (_Float16)(v[c][j] * inv[j]);
    *(v4h*)(p + base + (long)c * HW_) = h4;           // 8B aligned store
  }
}

// ---------------------------------------------------------------------------
// Kernel 2: vertical (H) 11-tap conv as ONE v_wmma_f32_16x16x32_f16 per
// 16x16 tile:  D(16x16) = A(16x32 banded kernel) x B(32x16 prob window) + 0.
//   A[m][k] = kern[k-m] (band 0..10), B[k][n] = p[h0-5+k][w0+n] (zero-pad rows)
// One wave per (b, c, h-tile, w-tile). EXEC all ones at the WMMA.
// ---------------------------------------------------------------------------
__global__ __launch_bounds__(256) void crf_conv_h(
    const _Float16* __restrict__ p, const float* __restrict__ spacings,
    _Float16* __restrict__ t) {
  int tid  = blockIdx.x * 8 + (threadIdx.x >> 5);     // tile id, exact fit
  int lane = threadIdx.x & 31;
  int g    = lane >> 4;
  int ln   = lane & 15;
  int wt = tid % WT_;
  int ht = (tid / WT_) % HT_;
  int c  = (tid / (WT_ * HT_)) % C_;
  int b  =  tid / (WT_ * HT_ * C_);
  int h0 = ht * 16, w0 = wt * 16;
  float s = spacings[2 * b + 0];                      // H-axis spacing
  long cb = ((long)b * C_ + c) * (long)HW_;

  // A fragment (banded kernel): contiguous k order K = e + 16g (B matches)
  v16h a;
#pragma unroll
  for (int e = 0; e < 16; ++e) {
    int K = e + 16 * g;
    a[e] = (_Float16)gauss_tap(K - ln, s);
  }
  // B fragment (prob window): contiguous k order K = e + 16g (as in A), N = ln
  v16h bm;
#pragma unroll
  for (int e = 0; e < 16; ++e) {
    int K = e + 16 * g;
    int r = h0 - 5 + K;
    _Float16 pv = (_Float16)0.0f;
    if (r >= 0 && r < H_) pv = p[cb + (long)r * W_ + (w0 + ln)];
    bm[e] = pv;
  }
  v8f acc = {};
  acc = __builtin_amdgcn_wmma_f32_16x16x32_f16(
      false, a, false, bm, (short)0, acc, false, false);
#pragma unroll
  for (int i = 0; i < 8; ++i) {
    int r = h0 + i + 8 * g;
    t[cb + (long)r * W_ + (w0 + ln)] = (_Float16)acc[i];
  }
}

// ---------------------------------------------------------------------------
// Kernel 3: horizontal (W) conv + unary add:
//   D(16x16) = A(16x32 data window of t) x B(32x16 banded kernel) + C(x0 tile)
// Window origin is w0-8 (not w0-5) so each lane's two 8-half A chunks are
// 16B-aligned: interior tiles (wt in [1,22]) load A with just TWO
// global_load_b128. Band becomes kern[K - n - 3]. Branch is wave-uniform.
// ---------------------------------------------------------------------------
__global__ __launch_bounds__(256) void crf_conv_w(
    const _Float16* __restrict__ t, const float* __restrict__ spacings,
    const float* __restrict__ x0, float* __restrict__ xout) {
  int tid  = blockIdx.x * 8 + (threadIdx.x >> 5);
  int lane = threadIdx.x & 31;
  int g    = lane >> 4;
  int ln   = lane & 15;
  int wt = tid % WT_;
  int ht = (tid / WT_) % HT_;
  int c  = (tid / (WT_ * HT_)) % C_;
  int b  =  tid / (WT_ * HT_ * C_);
  int h0 = ht * 16, w0 = wt * 16;
  float s = spacings[2 * b + 1];                      // W-axis spacing
  long cb = ((long)b * C_ + c) * (long)HW_;
  long rowbase = cb + (long)(h0 + ln) * W_;

  // A fragment: t[h0+ln][w0-8+K], K = e + 8g + (e>=8?8:0)
  //  -> elements 0..7  = halves at cols w0-8+8g  .. w0-1+8g   (contiguous)
  //  -> elements 8..15 = halves at cols w0+8+8g  .. w0+15+8g  (contiguous)
  v16h a;
  if (wt >= 1 && wt <= 22) {
    // 16B-aligned fast path: two b128 loads per lane
    const v8h lo = *(const v8h*)(t + rowbase + (w0 - 8 + 8 * g));
    const v8h hi = *(const v8h*)(t + rowbase + (w0 + 8 + 8 * g));
    a = __builtin_shufflevector(lo, hi, 0, 1, 2, 3, 4, 5, 6, 7,
                                        8, 9, 10, 11, 12, 13, 14, 15);
  } else {
    // edge tiles: guarded scalar loads (zero padding)
#pragma unroll
    for (int e = 0; e < 16; ++e) {
      int K   = e + 8 * g + ((e >= 8) ? 8 : 0);
      int col = w0 - 8 + K;
      _Float16 tv = (_Float16)0.0f;
      if (col >= 0 && col < W_) tv = t[rowbase + col];
      a[e] = tv;
    }
  }
  // B fragment (banded kernel, shifted origin): K = e + 8g + (e>=8?8:0), N = ln
  v16h bm;
#pragma unroll
  for (int e = 0; e < 16; ++e) {
    int K = e + 8 * g + ((e >= 8) ? 8 : 0);
    bm[e] = (_Float16)gauss_tap(K - ln - 3, s);
  }
  // C fragment: x0 tile (f32), row M = i + 8g, col N = ln
  v8f cacc;
#pragma unroll
  for (int i = 0; i < 8; ++i)
    cacc[i] = x0[cb + (long)(h0 + i + 8 * g) * W_ + (w0 + ln)];

  v8f d = __builtin_amdgcn_wmma_f32_16x16x32_f16(
      false, a, false, bm, (short)0, cacc, false, false);
#pragma unroll
  for (int i = 0; i < 8; ++i)
    xout[cb + (long)(h0 + i + 8 * g) * W_ + (w0 + ln)] = d[i];
}

// ---------------------------------------------------------------------------
// 5 iterations, 3 kernels each. Iteration 1 reads d_in[0]; x then lives in
// d_out (deterministic, graph-replay safe). p and t are f16 scratch in d_ws.
// ---------------------------------------------------------------------------
extern "C" void kernel_launch(void* const* d_in, const int* in_sizes, int n_in,
                              void* d_out, int out_size, void* d_ws, size_t ws_size,
                              hipStream_t stream) {
  const float* x0 = (const float*)d_in[0];
  const float* sp = (const float*)d_in[1];
  float* xout     = (float*)d_out;
  _Float16* p     = (_Float16*)d_ws;
  _Float16* t     = p + (size_t)NTOT_;

  dim3 blk(256);
  dim3 smGrid(((B_ * HW_) / 4 + 255) / 256);
  dim3 cvGrid(TILES_ / 8);                            // 8 waves (tiles) per block

  for (int it = 0; it < 5; ++it) {
    const float* xin = (it == 0) ? x0 : xout;
    crf_softmax_f16<<<smGrid, blk, 0, stream>>>(xin, p);
    crf_conv_h<<<cvGrid, blk, 0, stream>>>(p, sp, t);
    crf_conv_w<<<cvGrid, blk, 0, stream>>>(t, sp, x0, xout);
  }
}
